// InteractionModel_65721589563648
// MI455X (gfx1250) — matrix operand order repaired and checked by reading, not compiled, not run on hardware
//
#include <hip/hip_runtime.h>
#include <hip/hip_bf16.h>
#include <stdint.h>

typedef __attribute__((ext_vector_type(16))) __bf16 v16bf;
typedef __attribute__((ext_vector_type(8)))  float  v8f;
typedef __attribute__((ext_vector_type(2)))  float  v2f;

constexpr int kB1 = 128;   // images
constexpr int kS1 = 49;    // image tokens
constexpr int kB2 = 64;    // texts
constexpr int kS2 = 32;    // text tokens
constexpr int kE  = 512;   // embed dim
constexpr int kNP = 64;    // padded image-token dim (49 -> 64)
constexpr float kInvTemp = 1.0f / 0.11f;

struct U4x2 { uint4 a, b; };

static __device__ __forceinline__ unsigned short f2bf(float f) {
  unsigned u = __builtin_bit_cast(unsigned, f);
  u += 0x7FFFu + ((u >> 16) & 1u);   // round-to-nearest-even
  return (unsigned short)(u >> 16);
}

// ---- prep: text f32 -> bf16, row-major [t][q][e] ----
__global__ void cvt_text_bf16(const float* __restrict__ text,
                              unsigned short* __restrict__ tb) {
  int idx = blockIdx.x * 256 + threadIdx.x;   // kB2*kS2*kE total
  tb[idx] = f2bf(text[idx]);
}

// ---- prep: imageT bf16 [i][e][n], n padded 49->64 with zeros ----
__global__ void build_imageT_bf16(const float* __restrict__ img,
                                  unsigned short* __restrict__ it) {
  int idx = blockIdx.x * 256 + threadIdx.x;   // kB1*kE*kNP total
  int n = idx & (kNP - 1);
  int rest = idx >> 6;
  int e = rest & (kE - 1);
  int i = rest >> 9;
  float v = (n < kS1) ? img[((size_t)i * kS1 + n) * kE + e] : 0.0f;
  it[idx] = f2bf(v);
}

// ---- Gram matrices G[i][k][k'] = <image[i,k], image[i,k']> in f32 ----
__global__ void gram_kernel(const float* __restrict__ img,
                            float* __restrict__ G) {
  int i = blockIdx.x;
  const float* base = img + (size_t)i * kS1 * kE;
  for (int idx = threadIdx.x; idx < kS1 * kS1; idx += 256) {
    int k  = idx / kS1;
    int kp = idx % kS1;
    const float4* a = (const float4*)(base + (size_t)k  * kE);
    const float4* b = (const float4*)(base + (size_t)kp * kE);
    float4 acc4 = {0.f, 0.f, 0.f, 0.f};
#pragma unroll 4
    for (int e4 = 0; e4 < kE / 4; ++e4) {
      float4 av = a[e4], bv = b[e4];
      acc4.x += av.x * bv.x; acc4.y += av.y * bv.y;
      acc4.z += av.z * bv.z; acc4.w += av.w * bv.w;
    }
    G[(size_t)i * (kS1 * kS1) + idx] = (acc4.x + acc4.y) + (acc4.z + acc4.w);
  }
}

// ---- main: per (i,t) block ----
// phase 1: WMMA bf16  logits[32][64] = text[t] * imageT[i]
// phase 2: softmax + weighted-logit dot (numerator)
// phase 3: WMMA f32   H = E*G ; norm^2 = rowdot(E,H)*invZ^2 ; masked mean
__global__ void __launch_bounds__(256)
interaction_main(const unsigned short* __restrict__ tb,   // [kB2][kS2][kE] bf16
                 const unsigned short* __restrict__ it,   // [kB1][kE][kNP] bf16
                 const float* __restrict__ G,              // [kB1][kS1][kS1]
                 const int* __restrict__ masks,            // [kB2][kS2]
                 float* __restrict__ out)                  // [kB2][kB1]
{
  __shared__ float lds_logit[kS2 * kNP];   // 8 KB: logits, then E*H products
  __shared__ float lds_s[kS2 * kNP];       // 8 KB: unnormalized exp, zero-padded
  __shared__ float lds_G[kNP * kNP];       // 16 KB: Gram zero-padded to 64x64
  __shared__ float lds_red[kS2];

  const int i    = blockIdx.x;
  const int t    = blockIdx.y;
  const int tid  = threadIdx.x;
  const int wave = tid >> 5;
  const int lane = tid & 31;
  const int hi   = lane >> 4;     // lane half select
  const int l15  = lane & 15;
  const int m_tile = wave >> 2;   // 0..1  (q tile)
  const int n_tile = wave & 3;    // 0..3  (image-token tile)

  // stage zero-padded Gram into LDS (consumed after first barrier)
  {
    const float* gsrc = G + (size_t)i * (kS1 * kS1);
    for (int idx = tid; idx < kNP * kNP; idx += 256) {
      int k  = idx >> 6;
      int kp = idx & 63;
      lds_G[idx] = (k < kS1 && kp < kS1) ? gsrc[k * kS1 + kp] : 0.0f;
    }
  }

  // ---- phase 1: WMMA bf16 GEMM: logits[32][64] ----
  const unsigned short* abase =
      tb + ((size_t)(t * kS2 + m_tile * 16 + l15)) * kE + 8 * hi;
  const unsigned short* bbase =
      it + ((size_t)i * kE + 8 * hi) * kNP + n_tile * 16 + l15;

  v8f acc = {};
#pragma unroll 4
  for (int e0 = 0; e0 < kE; e0 += 32) {
    U4x2 au, bu;
    au.a = *(const uint4*)(abase + e0);            // K = e0+8h .. +7
    au.b = *(const uint4*)(abase + e0 + 16);       // K = e0+16+8h .. +7
#pragma unroll
    for (int s = 0; s < 16; ++s)
      ((unsigned short*)&bu)[s] = bbase[(size_t)(e0 + (s < 8 ? s : s + 8)) * kNP];
    v16bf av = __builtin_bit_cast(v16bf, au);
    v16bf bv = __builtin_bit_cast(v16bf, bu);
    acc = __builtin_amdgcn_wmma_f32_16x16x32_bf16(
        false, av, false, bv, (short)0, acc, false, false);
  }

  // scatter logits to LDS: lane=N (l15), VGPR r -> M = r + 8*hi
  {
    float* dst = lds_logit + (m_tile * 16 + 8 * hi) * kNP + (n_tile * 16 + l15);
#pragma unroll
    for (int r = 0; r < 8; ++r) dst[r * kNP] = acc[r];
  }
  __syncthreads();

  // ---- phase 2: softmax over k (8 threads per q row), fully unrolled ----
  const int q = tid >> 3;
  const int j = tid & 7;
  const float* row = lds_logit + q * kNP;

  float mx = -3.4e38f;
#pragma unroll
  for (int itn = 0; itn < 8; ++itn) {
    int k = j + itn * 8;
    float lg = row[k];
    mx = fmaxf(mx, (k < kS1) ? lg : -3.4e38f);
  }
#pragma unroll
  for (int m = 4; m >= 1; m >>= 1) mx = fmaxf(mx, __shfl_xor(mx, m, 8));

  float psum = 0.f, pdot = 0.f;
#pragma unroll
  for (int itn = 0; itn < 8; ++itn) {
    int k = j + itn * 8;
    float lg = row[k];
    float ex = (k < kS1) ? __expf((lg - mx) * kInvTemp) : 0.0f;
    lds_s[q * kNP + k] = ex;           // zero-padded beyond kS1
    psum += ex;
    pdot += ex * lg;
  }
#pragma unroll
  for (int m = 4; m >= 1; m >>= 1) {
    psum += __shfl_xor(psum, m, 8);
    pdot += __shfl_xor(pdot, m, 8);
  }
  const float invZ = 1.0f / psum;
  const float dot  = pdot * invZ;
  __syncthreads();   // lds_s complete; lds_logit reads done

  // ---- phase 3: H = E * G via f32 WMMA (K = 64 in steps of 4) ----
  // A (16x4 f32): lane = M (l15), K = 2*hi + {0,1}  -> contiguous pair
  // B (4x16 f32): VGPR j holds row K = 2*hi + j, N striped across lanes (l15)
  const float* arow = lds_s + (m_tile * 16 + l15) * kNP + 2 * hi;
  const float* bcol = lds_G + (2 * hi) * kNP + n_tile * 16 + l15;

  v8f hacc = {};
#pragma unroll
  for (int k0 = 0; k0 < kNP; k0 += 4) {
    v2f av2, bv2;
    av2.x = arow[k0];
    av2.y = arow[k0 + 1];
    bv2.x = bcol[(size_t)k0 * kNP];
    bv2.y = bcol[(size_t)(k0 + 1) * kNP];
    hacc = __builtin_amdgcn_wmma_f32_16x16x4_f32(
        false, av2, false, bv2, (short)0, hacc, false, false);
  }

  // products E[m][n]*H[m][n] overwrite lds_logit (logits dead after phase 2)
  {
    const int nn = n_tile * 16 + l15;
    const int mbase = m_tile * 16 + 8 * hi;
#pragma unroll
    for (int r = 0; r < 8; ++r) {
      int m = mbase + r;
      lds_logit[m * kNP + nn] = hacc[r] * lds_s[m * kNP + nn];
    }
  }
  __syncthreads();

  // norm^2 = (sum_n prod[q][n]) * invZ^2  (pads contribute exact zeros)
  float pn = 0.f;
#pragma unroll
  for (int nn = 0; nn < 8; ++nn) pn += lds_logit[q * kNP + j * 8 + nn];
#pragma unroll
  for (int m = 4; m >= 1; m >>= 1) pn += __shfl_xor(pn, m, 8);
  const float norm2 = pn * invZ * invZ;
  const float sim = dot / fmaxf(sqrtf(norm2), 1e-12f);

  if (j == 0) lds_red[q] = sim * (float)masks[t * kS2 + q];
  __syncthreads();
  if (tid == 0) {
    float s = 0.f, mn = 0.f;
    for (int qq = 0; qq < kS2; ++qq) {
      s  += lds_red[qq];
      mn += (float)masks[t * kS2 + qq];
    }
    out[(size_t)t * kB1 + i] = s / mn;
  }
}

extern "C" void kernel_launch(void* const* d_in, const int* in_sizes, int n_in,
                              void* d_out, int out_size, void* d_ws, size_t ws_size,
                              hipStream_t stream) {
  const float* image = (const float*)d_in[0];   // [128,49,512] f32
  const float* text  = (const float*)d_in[1];   // [64,32,512]  f32
  const int*   masks = (const int*)d_in[2];     // [64,32]      i32
  float* out = (float*)d_out;                   // [64,128]     f32

  char* ws = (char*)d_ws;
  unsigned short* tb = (unsigned short*)(ws);                            // 2 MB
  unsigned short* it = (unsigned short*)(ws + (size_t)2  * 1024 * 1024); // 8 MB
  float*          G  = (float*)         (ws + (size_t)10 * 1024 * 1024); // 1.2 MB

  hipLaunchKernelGGL(cvt_text_bf16, dim3((kB2 * kS2 * kE) / 256), dim3(256),
                     0, stream, text, tb);
  hipLaunchKernelGGL(build_imageT_bf16, dim3((kB1 * kE * kNP) / 256), dim3(256),
                     0, stream, image, it);
  hipLaunchKernelGGL(gram_kernel, dim3(kB1), dim3(256), 0, stream, image, G);
  hipLaunchKernelGGL(interaction_main, dim3(kB1, kB2), dim3(256), 0, stream,
                     tb, it, G, masks, out);
}
